// VJEPA2JointDiffuser_1176821039789
// MI455X (gfx1250) — matrix operand order repaired and checked by reading, compiled, not run on hardware
//
#include <hip/hip_runtime.h>
#include <hip/hip_bf16.h>

#define HID 1024
#define NH  16
#define DH  64
#define NV  2048
#define NA  64
#define NTOT 2112
#define BATCH 2
#define SCALE 0.125f

typedef __bf16 bf16;
typedef __attribute__((ext_vector_type(16))) __bf16 v16bf;
typedef __attribute__((ext_vector_type(8)))  __bf16 v8bf;
typedef __attribute__((ext_vector_type(8)))  float  v8f;
typedef __attribute__((ext_vector_type(4)))  int    i32x4;

#if __has_builtin(__builtin_amdgcn_global_load_async_to_lds_b128) && \
    __has_builtin(__builtin_amdgcn_s_wait_asynccnt)
#define USE_ASYNC_LDS 1
#else
#define USE_ASYNC_LDS 0
#endif

#if USE_ASYNC_LDS
// global -> LDS async copy of 16 bytes per lane (tracked by ASYNCcnt)
__device__ __forceinline__ void async_ld_lds_b128(const void* g, void* l) {
    __builtin_amdgcn_global_load_async_to_lds_b128(
        (__attribute__((address_space(1))) i32x4*)(
            (__attribute__((address_space(1))) void*)(void*)g),
        (__attribute__((address_space(3))) i32x4*)(
            (__attribute__((address_space(3))) void*)(void*)l),
        0, 0);
}
#endif

// ---------------------------------------------------------------- helpers
__device__ __forceinline__ float half_red_max(float v) {
    v = fmaxf(v, __shfl_xor(v, 1, 32));
    v = fmaxf(v, __shfl_xor(v, 2, 32));
    v = fmaxf(v, __shfl_xor(v, 4, 32));
    v = fmaxf(v, __shfl_xor(v, 8, 32));
    return v;   // max over the 16 lanes of this wave-half
}
__device__ __forceinline__ float half_red_sum(float v) {
    v += __shfl_xor(v, 1, 32);
    v += __shfl_xor(v, 2, 32);
    v += __shfl_xor(v, 4, 32);
    v += __shfl_xor(v, 8, 32);
    return v;
}

// ---------------------------------------------------------------- fp32 -> bf16
__global__ void cvt_f32_bf16(const float* __restrict__ src, bf16* __restrict__ dst, int n) {
    int i = (blockIdx.x * blockDim.x + threadIdx.x) * 4;
    if (i >= n) return;
    float4 v = *(const float4*)(src + i);
    dst[i + 0] = (bf16)v.x;
    dst[i + 1] = (bf16)v.y;
    dst[i + 2] = (bf16)v.z;
    dst[i + 3] = (bf16)v.w;
}

// gather ctx[:, NV:, :] (fp32 in d_out) into a compact bf16 buffer
__global__ void gather_action_bf16(const float* __restrict__ ctx, bf16* __restrict__ dst) {
    int i = blockIdx.x * blockDim.x + threadIdx.x;   // 0 .. BATCH*NA*HID
    if (i >= BATCH * NA * HID) return;
    int o = i & (HID - 1);
    int t = (i >> 10) & (NA - 1);
    int b = i >> 16;
    dst[i] = (bf16)ctx[((size_t)(b * NTOT + NV + t) << 10) + o];
}

// ---------------------------------------------------------------- NT GEMM: OUT = X @ W^T + bias
// X: [rows, HID] bf16 row-major.  W: [HID, HID] bf16 row-major (out_dim major).
// Wave computes a 32x64 tile: 2 A-fragments x 4 B-fragments -> 8 WMMAs per k-chunk.
// out_row = (r / tok_in) * tok_out_stride + tok_base + (r % tok_in)
__global__ void gemm_nt_bf16(const bf16* __restrict__ X, const bf16* __restrict__ W,
                             const float* __restrict__ bias, float* __restrict__ OUT,
                             int tok_in, int tok_out_stride, int tok_base) {
    const int lane = threadIdx.x & 31;
    const int wave = threadIdx.x >> 5;
    const int hl   = lane >> 4;
    const int ln   = lane & 15;
    const int row0 = blockIdx.x * 32;
    const int o0   = blockIdx.y * 256 + wave * 64;

    v8f acc[2][4];
#pragma unroll
    for (int g = 0; g < 2; ++g)
#pragma unroll
        for (int f = 0; f < 4; ++f) acc[g][f] = v8f{};

    const bf16* xrow0 = X + (size_t)(row0 + ln) * HID;
    const bf16* xrow1 = X + (size_t)(row0 + 16 + ln) * HID;
    for (int k0 = 0; k0 < HID; k0 += 32) {
        // A fragments: lane holds row; k = 16*hl + j
        v16bf a0 = *(const v16bf*)(xrow0 + k0 + 16 * hl);
        v16bf a1 = *(const v16bf*)(xrow1 + k0 + 16 * hl);
#pragma unroll
        for (int f = 0; f < 4; ++f) {
            // B fragment: b[j] = W[o0+f*16+ln][k0 + 16*hl + j]  (contiguous 32B)
            v16bf b = *(const v16bf*)(W + (size_t)(o0 + f * 16 + ln) * HID + k0 + 16 * hl);
            acc[0][f] = __builtin_amdgcn_wmma_f32_16x16x32_bf16(
                false, a0, false, b, (short)0, acc[0][f], false, false);
            acc[1][f] = __builtin_amdgcn_wmma_f32_16x16x32_bf16(
                false, a1, false, b, (short)0, acc[1][f], false, false);
        }
    }
#pragma unroll
    for (int g = 0; g < 2; ++g)
#pragma unroll
        for (int f = 0; f < 4; ++f) {
            int o = o0 + f * 16 + ln;
            float bv = bias[o];
#pragma unroll
            for (int e = 0; e < 8; ++e) {
                int r    = row0 + 16 * g + e + 8 * hl;
                int orow = (r / tok_in) * tok_out_stride + tok_base + (r % tok_in);
                OUT[(size_t)orow * HID + o] = acc[g][f][e] + bv;
            }
        }
}

// ---------------------------------------------------------------- RoPE (VJEPA2 style) + cvt to bf16
__global__ void rope_cvt_bf16(const float* __restrict__ src, bf16* __restrict__ dst) {
    int idx = blockIdx.x * blockDim.x + threadIdx.x;   // (b, token, head)
    if (idx >= BATCH * NTOT * NH) return;
    int h = idx % NH;
    int t = (idx / NH) % NTOT;
    int b = idx / (NH * NTOT);

    float pos[3];
    if (t < NV) {
        pos[0] = (float)(t / 256);
        pos[1] = (float)((t % 256) / 16);
        pos[2] = (float)(t % 16);
    } else {                           // action tokens: linspace(0,0,NA) == 0
        pos[0] = pos[1] = pos[2] = 0.f;
    }
    const float* s = src + (size_t)(b * NTOT + t) * HID + h * DH;
    bf16*        d = dst + (size_t)(b * NTOT + t) * HID + h * DH;

    float x[DH], y[DH];
#pragma unroll
    for (int i = 0; i < DH; ++i) x[i] = s[i];
#pragma unroll
    for (int axis = 0; axis < 3; ++axis) {
        int base = axis * 20;
        float p = pos[axis];
#pragma unroll
        for (int pr = 0; pr < 10; ++pr) {
            int t0 = 2 * pr, t1 = 2 * pr + 1;
            float om0 = __powf(10000.f, -(float)(t0 % 10) * 0.1f);
            float om1 = __powf(10000.f, -(float)(t1 % 10) * 0.1f);
            float f0 = p * om0, f1 = p * om1;
            float x0 = x[base + t0], x1 = x[base + t1];
            y[base + t0] = x0 * __cosf(f0) - x1 * __sinf(f0);
            y[base + t1] = x1 * __cosf(f1) + x0 * __sinf(f1);
        }
    }
    y[60] = x[60]; y[61] = x[61]; y[62] = x[62]; y[63] = x[63];
#pragma unroll
    for (int i = 0; i < DH; ++i) d[i] = (bf16)y[i];
}

// ---------------------------------------------------------------- flash attention
// block = 128 threads (4 waves); wave handles 16 queries; block loops over 32-token K/V chunks.
__global__ void attn_flash(const bf16* __restrict__ Q, const bf16* __restrict__ K,
                           const bf16* __restrict__ V, float* __restrict__ OUT) {
    __shared__ __align__(64) bf16 Kl[32 * 64];       // [tok][dim]
    __shared__ __align__(64) bf16 Vt[64 * 32];       // [dim][tok]  (transposed)
    __shared__ __align__(64) bf16 Pl[4][16 * 32];    // per-wave P tile, [row][col]

    const int lane = threadIdx.x & 31;
    const int wave = threadIdx.x >> 5;
    const int hl   = lane >> 4;
    const int ln   = lane & 15;
    const int b    = blockIdx.z;
    const int h    = blockIdx.y;
    const int q0   = (blockIdx.x * 4 + wave) * 16;

    const size_t bh = (size_t)b * NTOT * HID + (size_t)h * DH;

    // Q fragments for the two K-dim=32 slabs of DH=64
    v16bf qa0, qa1;
    {
        const bf16* qr = Q + bh + (size_t)(q0 + ln) * HID;
        qa0 = *(const v16bf*)(qr + 16 * hl);
        qa1 = *(const v16bf*)(qr + 32 + 16 * hl);
    }

    float mrow[8], lrow[8];
#pragma unroll
    for (int e = 0; e < 8; ++e) { mrow[e] = -__builtin_inff(); lrow[e] = 0.f; }
    v8f o_acc[4] = {v8f{}, v8f{}, v8f{}, v8f{}};

    for (int c0 = 0; c0 < NTOT; c0 += 32) {
        // ---- K chunk -> LDS (async DMA path when available), V chunk -> LDS transposed
        {
            int e0  = threadIdx.x * 16;          // 128 thr * 16 elems = 2048 elems (4KB)
            int tok = e0 >> 6, dd = e0 & 63;
            const bf16* kp = K + bh + (size_t)(c0 + tok) * HID + dd;
#if USE_ASYNC_LDS
            async_ld_lds_b128(kp,     Kl + e0);
            async_ld_lds_b128(kp + 8, Kl + e0 + 8);
#else
            *(v8bf*)(Kl + e0)     = *(const v8bf*)kp;
            *(v8bf*)(Kl + e0 + 8) = *(const v8bf*)(kp + 8);
#endif
            int tv = threadIdx.x >> 2;           // 0..31 token
            int dv = (threadIdx.x & 3) * 16;     // 0,16,32,48
            const bf16* vp = V + bh + (size_t)(c0 + tv) * HID + dv;
            v8bf v0 = *(const v8bf*)vp;
            v8bf v1 = *(const v8bf*)(vp + 8);
#pragma unroll
            for (int j = 0; j < 8; ++j) {
                Vt[(dv + j) * 32 + tv]     = v0[j];
                Vt[(dv + 8 + j) * 32 + tv] = v1[j];
            }
        }
#if USE_ASYNC_LDS
        __builtin_amdgcn_s_wait_asynccnt(0);
#endif
        __syncthreads();

        // ---- S = Q K^T for two 16-token subtiles
        v8f s[2];
#pragma unroll
        for (int t = 0; t < 2; ++t) {
            const bf16* kr = Kl + (16 * t + ln) * 64;
            v16bf kb0 = *(const v16bf*)(kr + 16 * hl);        // i = 16*hl + j
            v16bf kb1 = *(const v16bf*)(kr + 32 + 16 * hl);
            v8f c = {};
            c = __builtin_amdgcn_wmma_f32_16x16x32_bf16(false, qa0, false, kb0, (short)0, c, false, false);
            c = __builtin_amdgcn_wmma_f32_16x16x32_bf16(false, qa1, false, kb1, (short)0, c, false, false);
            s[t] = c;
        }

        // ---- online softmax (row = element e + 8*hl, cols in lanes of half)
#pragma unroll
        for (int e = 0; e < 8; ++e) {
            float v0 = s[0][e] * SCALE, v1 = s[1][e] * SCALE;
            float mx = half_red_max(fmaxf(v0, v1));
            float m2 = fmaxf(mrow[e], mx);
            float corr = __expf(mrow[e] - m2);
            mrow[e] = m2;
            lrow[e] *= corr;
#pragma unroll
            for (int f = 0; f < 4; ++f) o_acc[f][e] *= corr;
            float p0 = __expf(v0 - m2), p1 = __expf(v1 - m2);
            s[0][e] = p0; s[1][e] = p1;
            lrow[e] += half_red_sum(p0 + p1);
        }

        // ---- relayout P (C-layout -> A-layout) through per-wave LDS
        bf16* pw = Pl[wave];
#pragma unroll
        for (int t = 0; t < 2; ++t)
#pragma unroll
            for (int e = 0; e < 8; ++e)
                pw[(e + 8 * hl) * 32 + 16 * t + ln] = (bf16)s[t][e];
        asm volatile("s_wait_dscnt 0" ::: "memory");

        v16bf pa = *(const v16bf*)(pw + ln * 32 + 16 * hl);

        // ---- O += P @ V  (B frag: b[j] = V[tok=j+16*hl][d=f*16+ln] from Vt, contiguous)
#pragma unroll
        for (int f = 0; f < 4; ++f) {
            v16bf vb = *(const v16bf*)(Vt + (f * 16 + ln) * 32 + 16 * hl);
            o_acc[f] = __builtin_amdgcn_wmma_f32_16x16x32_bf16(
                false, pa, false, vb, (short)0, o_acc[f], false, false);
        }
        __syncthreads();
    }

    // ---- normalize + store fp32 ctx
    float inv[8];
#pragma unroll
    for (int e = 0; e < 8; ++e) inv[e] = 1.f / lrow[e];
#pragma unroll
    for (int f = 0; f < 4; ++f)
#pragma unroll
        for (int e = 0; e < 8; ++e)
            OUT[((size_t)b * NTOT + q0 + e + 8 * hl) * HID + h * DH + f * 16 + ln] =
                o_acc[f][e] * inv[e];
}

// ---------------------------------------------------------------- launch
extern "C" void kernel_launch(void* const* d_in, const int* in_sizes, int n_in,
                              void* d_out, int out_size, void* d_ws, size_t ws_size,
                              hipStream_t stream) {
    const float* xv  = (const float*)d_in[0];
    const float* xa  = (const float*)d_in[1];
    const float* Wf[7]  = {(const float*)d_in[2],  (const float*)d_in[4],  (const float*)d_in[6],
                           (const float*)d_in[8],  (const float*)d_in[10], (const float*)d_in[12],
                           (const float*)d_in[14]};                 // Wq Wk Wv Wqa Wka Wva Wpa
    const float* bias[7] = {(const float*)d_in[3],  (const float*)d_in[5],  (const float*)d_in[7],
                            (const float*)d_in[9],  (const float*)d_in[11], (const float*)d_in[13],
                            (const float*)d_in[15]};

    char* ws = (char*)d_ws;
    size_t off = 0;
    auto alloc = [&](size_t bytes) -> void* {
        off = (off + 255) & ~(size_t)255;
        void* p = ws + off;
        off += bytes;
        return p;
    };

    const int nXV = BATCH * NV * HID, nXA = BATCH * NA * HID;
    const int nW  = HID * HID;
    const int nQK = BATCH * NTOT * HID;

    bf16* xv_bf = (bf16*)alloc((size_t)nXV * 2);
    bf16* xa_bf = (bf16*)alloc((size_t)nXA * 2);
    bf16* w_bf[7];
    for (int i = 0; i < 7; ++i) w_bf[i] = (bf16*)alloc((size_t)nW * 2);
    float* qf = (float*)alloc((size_t)nQK * 4);
    float* kf = (float*)alloc((size_t)nQK * 4);
    float* vf = (float*)alloc((size_t)nQK * 4);
    bf16* qb = (bf16*)alloc((size_t)nQK * 2);
    bf16* kb = (bf16*)alloc((size_t)nQK * 2);
    bf16* vb = (bf16*)alloc((size_t)nQK * 2);
    bf16* ca_bf = (bf16*)alloc((size_t)nXA * 2);

    auto cvt = [&](const float* s, bf16* d, int n) {
        cvt_f32_bf16<<<(n / 4 + 255) / 256, 256, 0, stream>>>(s, d, n);
    };
    cvt(xv, xv_bf, nXV);
    cvt(xa, xa_bf, nXA);
    for (int i = 0; i < 7; ++i) cvt(Wf[i], w_bf[i], nW);

    // QKV projections -> fp32 [B, 2112, 1024]
    dim3 blk(128);
    dim3 gv(BATCH * NV / 32, HID / 256);
    dim3 ga(BATCH * NA / 32, HID / 256);
    gemm_nt_bf16<<<gv, blk, 0, stream>>>(xv_bf, w_bf[0], bias[0], qf, NV, NTOT, 0);
    gemm_nt_bf16<<<gv, blk, 0, stream>>>(xv_bf, w_bf[1], bias[1], kf, NV, NTOT, 0);
    gemm_nt_bf16<<<gv, blk, 0, stream>>>(xv_bf, w_bf[2], bias[2], vf, NV, NTOT, 0);
    gemm_nt_bf16<<<ga, blk, 0, stream>>>(xa_bf, w_bf[3], bias[3], qf, NA, NTOT, NV);
    gemm_nt_bf16<<<ga, blk, 0, stream>>>(xa_bf, w_bf[4], bias[4], kf, NA, NTOT, NV);
    gemm_nt_bf16<<<ga, blk, 0, stream>>>(xa_bf, w_bf[5], bias[5], vf, NA, NTOT, NV);

    // RoPE on Q,K; plain convert V
    int nrope = BATCH * NTOT * NH;
    rope_cvt_bf16<<<(nrope + 255) / 256, 256, 0, stream>>>(qf, qb);
    rope_cvt_bf16<<<(nrope + 255) / 256, 256, 0, stream>>>(kf, kb);
    cvt(vf, vb, nQK);

    // attention -> fp32 ctx in d_out[0 .. B*NTOT*HID)
    float* ctx = (float*)d_out;
    attn_flash<<<dim3(NTOT / 64, NH, BATCH), blk, 0, stream>>>(qb, kb, vb, ctx);

    // action output projection -> d_out tail
    gather_action_bf16<<<(nXA + 255) / 256, 256, 0, stream>>>(ctx, ca_bf);
    float* out_a = ctx + (size_t)BATCH * NTOT * HID;
    gemm_nt_bf16<<<dim3(BATCH * NA / 32, HID / 256), blk, 0, stream>>>(
        ca_bf, w_bf[6], bias[6], out_a, NA, NA, 0);
}
